// RelCls_33741263077663
// MI455X (gfx1250) — matrix operand order repaired and checked by reading, not compiled, not run on hardware
//
#include <hip/hip_runtime.h>
#include <hip/hip_bf16.h>

typedef __attribute__((ext_vector_type(16))) _Float16 v16h;
typedef __attribute__((ext_vector_type(8)))  _Float16 v8h;
typedef __attribute__((ext_vector_type(8)))  float    v8f;
typedef __attribute__((ext_vector_type(4)))  float    v4f;

#define NN 100000
#define EE 800000
#define HH 128
#define CC 26

// f16-element offsets inside d_ws for packed B-fragments
#define WOFF_WE   0
#define WOFF_WQ   32768
#define WOFF_WK   65536
#define WOFF_WV   98304
#define WOFF_WC1  131072
#define WOFF_WC2  139264
#define WOFF_WC3  141312

#define WMMA_F16(A,B,C) __builtin_amdgcn_wmma_f32_16x16x32_f16(false,(A),false,(B),(short)0,(C),false,false)

// ---------------------------------------------------------------------------
// Pack a row-major f32 weight [K][Nsrc] into f16 WMMA B-fragments:
// layout [kc][nt][lane][16 halfs]; h=lane>>4: K=kc*32+8h+j (j<8), K=kc*32+16+8h+(j-8) (j>=8),
// N = nt*16 + (lane&15). Zero-pads (used for Wc3 N=26 -> 32).
// ---------------------------------------------------------------------------
__global__ void pack_b(const float* __restrict__ src, _Float16* __restrict__ dst,
                       int K, int Nsrc, int nkc, int nnt) {
  int i = blockIdx.x * blockDim.x + threadIdx.x;
  int total = nkc * nnt * 32 * 16;
  if (i >= total) return;
  int j    = i & 15;
  int lane = (i >> 4) & 31;
  int nt   = (i >> 9) % nnt;
  int kc   = (i >> 9) / nnt;
  int n = nt * 16 + (lane & 15);
  int k = kc * 32 + ((lane < 16) ? 0 : 8) + ((j < 8) ? j : j + 8);
  float v = 0.0f;
  if (n < Nsrc && k < K) v = src[k * Nsrc + n];
  dst[i] = (_Float16)v;
}

// ---------------------------------------------------------------------------
// Main fused kernel: 1 block (256 thr = 8 waves) handles 32 edges.
// ---------------------------------------------------------------------------
#define SMEM_BYTES 33024
#define OFF_A      0        // f16 A-frags [mat2][msub2][kc8][lane32][16]  (32768 B)
#define OFF_QK     0        // f32 [32][132]  (aliases A after barrier)
#define OFF_STATS  16896    // f32 rowmax[32], rowinv[32]
#define OFF_FUSED  17152    // f16 [32][136]
#define OFF_H1     25856    // f16 [32][72]
#define OFF_H2     30464    // f16 [32][40]

__launch_bounds__(256)
__global__ void relcls_main(const float* __restrict__ nf, const float* __restrict__ dsc,
                            const int* __restrict__ sub, const int* __restrict__ obj,
                            const float* __restrict__ be, const float* __restrict__ bq,
                            const float* __restrict__ bk, const float* __restrict__ bv,
                            const float* __restrict__ bc1, const float* __restrict__ bc2,
                            const float* __restrict__ bc3,
                            const _Float16* __restrict__ wpack,
                            float* __restrict__ out) {
  __shared__ __align__(16) char smem[SMEM_BYTES];
  const int tid  = threadIdx.x;
  const int lane = tid & 31;
  const int w    = tid >> 5;           // wave id = N-tile for main GEMMs
  const int tile = blockIdx.x;         // 32 edges per block (E % 32 == 0)

  // ---- Phase A: gather endpoint rows, convert f32->f16, store as A-frags ----
  {
    _Float16* ldsA = (_Float16*)(smem + OFF_A);
#pragma unroll
    for (int it = 0; it < 8; ++it) {
      int tsk = tid + 256 * it;        // 2048 tasks: 2 mats x 32 edges x 32 segs
      int mat = tsk >> 10;             // 0 = geo(node_feat), 1 = nongeo(descriptor)
      int rem = tsk & 1023;
      int m   = rem >> 5;              // edge row 0..31
      int seg = rem & 31;              // 8-float segment of the 256-wide row
      int e = tile * 32 + m;
      int node = (seg < 16) ? sub[e] : obj[e];
      const float* src = ((mat == 0) ? nf : dsc) + node * HH + (seg & 15) * 8;
      v4f a0 = *(const v4f*)(src);
      v4f a1 = *(const v4f*)(src + 4);
      int kc = seg >> 2;
      int o  = (seg & 3) * 8;          // K offset within 32-chunk: 0,8,16,24
      int flane = (m & 15) + ((o == 8 || o == 24) ? 16 : 0);
      int hoff  = (o >= 16) ? 8 : 0;
      int msub  = m >> 4;
      v8h h = { (_Float16)a0[0], (_Float16)a0[1], (_Float16)a0[2], (_Float16)a0[3],
                (_Float16)a1[0], (_Float16)a1[1], (_Float16)a1[2], (_Float16)a1[3] };
      *(v8h*)(ldsA + ((((mat * 2 + msub) * 8 + kc) * 32 + flane) * 16 + hoff)) = h;
    }
  }
  __syncthreads();

  // ---- Phase B: four GEMMs (origin/query/key/value), K=256, wave owns 16 cols ----
  v8f accO0 = {}, accO1 = {}, accQ0 = {}, accQ1 = {};
  v8f accK0 = {}, accK1 = {}, accV0 = {}, accV1 = {};
  {
    const _Float16* ldsA = (const _Float16*)(smem + OFF_A);
#pragma unroll
    for (int kc = 0; kc < 8; ++kc) {
      int bidx = ((kc * 8 + w) * 32 + lane) * 16;
      v16h bWe = *(const v16h*)(wpack + WOFF_WE + bidx);
      v16h bWq = *(const v16h*)(wpack + WOFF_WQ + bidx);
      v16h bWk = *(const v16h*)(wpack + WOFF_WK + bidx);
      v16h bWv = *(const v16h*)(wpack + WOFF_WV + bidx);
      v16h ag0 = *(const v16h*)(ldsA + (((     kc) * 32 + lane) * 16));  // geo,    msub0
      v16h an0 = *(const v16h*)(ldsA + (((16 + kc) * 32 + lane) * 16));  // nongeo, msub0
      accO0 = WMMA_F16(ag0, bWe, accO0);
      accQ0 = WMMA_F16(ag0, bWq, accQ0);
      accK0 = WMMA_F16(an0, bWk, accK0);
      accV0 = WMMA_F16(an0, bWv, accV0);
      v16h ag1 = *(const v16h*)(ldsA + ((( 8 + kc) * 32 + lane) * 16));  // geo,    msub1
      v16h an1 = *(const v16h*)(ldsA + (((24 + kc) * 32 + lane) * 16));  // nongeo, msub1
      accO1 = WMMA_F16(ag1, bWe, accO1);
      accQ1 = WMMA_F16(ag1, bWq, accQ1);
      accK1 = WMMA_F16(an1, bWk, accK1);
      accV1 = WMMA_F16(an1, bWv, accV1);
    }
  }
  __syncthreads();   // A-frag region is now dead; reuse for q*k

  // ---- Phase C: add biases, t = (q+bq)*(k+bk) -> LDS; keep biased origin/value ----
  float* qk = (float*)(smem + OFF_QK);
  {
    int col = w * 16 + (lane & 15);
    float vbe = be[col], vbq = bq[col], vbk = bk[col], vbv = bv[col];
    int hb = (lane >> 4) << 3;
#pragma unroll
    for (int r = 0; r < 8; ++r) {
      float t0 = (accQ0[r] + vbq) * (accK0[r] + vbk);
      accQ0[r] = t0;
      accO0[r] += vbe; accV0[r] += vbv;
      qk[(r + hb) * 132 + col] = t0;
      float t1 = (accQ1[r] + vbq) * (accK1[r] + vbk);
      accQ1[r] = t1;
      accO1[r] += vbe; accV1[r] += vbv;
      qk[(16 + r + hb) * 132 + col] = t1;
    }
  }
  __syncthreads();

  // ---- Phase D: per-row softmax stats over 128 features (8 threads / row) ----
  float* rowmax = (float*)(smem + OFF_STATS);
  float* rowinv = rowmax + 32;
  {
    int row = tid >> 3;
    int c0  = (tid & 7) * 16;
    const float* rp = qk + row * 132 + c0;
    v4f x0 = *(const v4f*)(rp);
    v4f x1 = *(const v4f*)(rp + 4);
    v4f x2 = *(const v4f*)(rp + 8);
    v4f x3 = *(const v4f*)(rp + 12);
    float mx = x0[0];
#pragma unroll
    for (int i = 1; i < 4; ++i) mx = fmaxf(mx, x0[i]);
#pragma unroll
    for (int i = 0; i < 4; ++i) { mx = fmaxf(mx, x1[i]); mx = fmaxf(mx, x2[i]); mx = fmaxf(mx, x3[i]); }
    mx = fmaxf(mx, __shfl_xor(mx, 1));
    mx = fmaxf(mx, __shfl_xor(mx, 2));
    mx = fmaxf(mx, __shfl_xor(mx, 4));
    float s = 0.0f;
#pragma unroll
    for (int i = 0; i < 4; ++i)
      s += __expf(x0[i] - mx) + __expf(x1[i] - mx) + __expf(x2[i] - mx) + __expf(x3[i] - mx);
    s += __shfl_xor(s, 1);
    s += __shfl_xor(s, 2);
    s += __shfl_xor(s, 4);
    if ((tid & 7) == 0) { rowmax[row] = mx; rowinv[row] = 1.0f / s; }
  }
  __syncthreads();

  // ---- Phase E: alpha = softmax(t), fused = value*alpha + origin -> f16 LDS ----
  _Float16* fused = (_Float16*)(smem + OFF_FUSED);
  {
    int col = w * 16 + (lane & 15);
    int hb  = (lane >> 4) << 3;
#pragma unroll
    for (int r = 0; r < 8; ++r) {
      int row0 = r + hb;
      float a0 = __expf(accQ0[r] - rowmax[row0]) * rowinv[row0];
      fused[row0 * 136 + col] = (_Float16)(accV0[r] * a0 + accO0[r]);
      int row1 = 16 + r + hb;
      float a1 = __expf(accQ1[r] - rowmax[row1]) * rowinv[row1];
      fused[row1 * 136 + col] = (_Float16)(accV1[r] * a1 + accO1[r]);
    }
  }
  __syncthreads();

  // ---- Phase F: h1 = relu(fused @ Wc1 + bc1)  [32x128]x[128x64], 8 wave-jobs ----
  _Float16* h1 = (_Float16*)(smem + OFF_H1);
  {
    int nt = w & 3, mj = w >> 2;
    int rowA = mj * 16 + (lane & 15);
    v8f acc = {};
#pragma unroll
    for (int kc = 0; kc < 4; ++kc) {
      int c0 = kc * 32 + ((lane >> 4) << 3);
      v8h lo = *(const v8h*)(fused + rowA * 136 + c0);
      v8h hi = *(const v8h*)(fused + rowA * 136 + c0 + 16);
      v16h a = __builtin_shufflevector(lo, hi, 0,1,2,3,4,5,6,7,8,9,10,11,12,13,14,15);
      v16h b = *(const v16h*)(wpack + WOFF_WC1 + ((kc * 4 + nt) * 32 + lane) * 16);
      acc = WMMA_F16(a, b, acc);
    }
    float bias = bc1[nt * 16 + (lane & 15)];
    int hb = (lane >> 4) << 3;
#pragma unroll
    for (int r = 0; r < 8; ++r)
      h1[(mj * 16 + r + hb) * 72 + nt * 16 + (lane & 15)] = (_Float16)fmaxf(acc[r] + bias, 0.0f);
  }
  __syncthreads();

  // ---- Phase G: h2 = h1 @ Wc2 + bc2 (no relu)  [32x64]x[64x32], 4 wave-jobs ----
  _Float16* h2 = (_Float16*)(smem + OFF_H2);
  if (w < 4) {
    int nt = w & 1, mj = w >> 1;
    int rowA = mj * 16 + (lane & 15);
    v8f acc = {};
#pragma unroll
    for (int kc = 0; kc < 2; ++kc) {
      int c0 = kc * 32 + ((lane >> 4) << 3);
      v8h lo = *(const v8h*)(h1 + rowA * 72 + c0);
      v8h hi = *(const v8h*)(h1 + rowA * 72 + c0 + 16);
      v16h a = __builtin_shufflevector(lo, hi, 0,1,2,3,4,5,6,7,8,9,10,11,12,13,14,15);
      v16h b = *(const v16h*)(wpack + WOFF_WC2 + ((kc * 2 + nt) * 32 + lane) * 16);
      acc = WMMA_F16(a, b, acc);
    }
    float bias = bc2[nt * 16 + (lane & 15)];
    int hb = (lane >> 4) << 3;
#pragma unroll
    for (int r = 0; r < 8; ++r)
      h2[(mj * 16 + r + hb) * 40 + nt * 16 + (lane & 15)] = (_Float16)(acc[r] + bias);
  }
  __syncthreads();

  // ---- Phase H: out = h2 @ Wc3 + bc3  [32x32]x[32x26(pad32)], 4 wave-jobs ----
  if (w < 4) {
    int nt = w & 1, mj = w >> 1;
    int rowA = mj * 16 + (lane & 15);
    int c0 = (lane >> 4) << 3;
    v8h lo = *(const v8h*)(h2 + rowA * 40 + c0);
    v8h hi = *(const v8h*)(h2 + rowA * 40 + c0 + 16);
    v16h a = __builtin_shufflevector(lo, hi, 0,1,2,3,4,5,6,7,8,9,10,11,12,13,14,15);
    v16h b = *(const v16h*)(wpack + WOFF_WC3 + (nt * 32 + lane) * 16);
    v8f acc = {};
    acc = WMMA_F16(a, b, acc);
    int col = nt * 16 + (lane & 15);
    if (col < CC) {
      float bias = bc3[col];
      int hb = (lane >> 4) << 3;
#pragma unroll
      for (int r = 0; r < 8; ++r) {
        int e = tile * 32 + mj * 16 + r + hb;
        out[e * CC + col] = acc[r] + bias;
      }
    }
  }
}

// ---------------------------------------------------------------------------
extern "C" void kernel_launch(void* const* d_in, const int* in_sizes, int n_in,
                              void* d_out, int out_size, void* d_ws, size_t ws_size,
                              hipStream_t stream) {
  const float* nf  = (const float*)d_in[0];
  const float* dsc = (const float*)d_in[1];
  const int*   sub = (const int*)d_in[2];
  const int*   obj = (const int*)d_in[3];
  const float* We  = (const float*)d_in[4];
  const float* be  = (const float*)d_in[5];
  const float* Wq  = (const float*)d_in[6];
  const float* bq  = (const float*)d_in[7];
  const float* Wk  = (const float*)d_in[8];
  const float* bk  = (const float*)d_in[9];
  const float* Wv  = (const float*)d_in[10];
  const float* bv  = (const float*)d_in[11];
  const float* Wc1 = (const float*)d_in[12];
  const float* bc1 = (const float*)d_in[13];
  const float* Wc2 = (const float*)d_in[14];
  const float* bc2 = (const float*)d_in[15];
  const float* Wc3 = (const float*)d_in[16];
  const float* bc3 = (const float*)d_in[17];
  _Float16* wp = (_Float16*)d_ws;

  // Pack weights into f16 WMMA B-fragment layout (deterministic, every call).
  pack_b<<<128, 256, 0, stream>>>(We,  wp + WOFF_WE,  256, 128, 8, 8);
  pack_b<<<128, 256, 0, stream>>>(Wq,  wp + WOFF_WQ,  256, 128, 8, 8);
  pack_b<<<128, 256, 0, stream>>>(Wk,  wp + WOFF_WK,  256, 128, 8, 8);
  pack_b<<<128, 256, 0, stream>>>(Wv,  wp + WOFF_WV,  256, 128, 8, 8);
  pack_b<<<32,  256, 0, stream>>>(Wc1, wp + WOFF_WC1, 128,  64, 4, 4);
  pack_b<<<8,   256, 0, stream>>>(Wc2, wp + WOFF_WC2,  64,  32, 2, 2);
  pack_b<<<4,   256, 0, stream>>>(Wc3, wp + WOFF_WC3,  32,  26, 1, 2);

  relcls_main<<<EE / 32, 256, 0, stream>>>(nf, dsc, sub, obj,
                                           be, bq, bk, bv, bc1, bc2, bc3,
                                           wp, (float*)d_out);
}
